// BIM_45724221833574
// MI455X (gfx1250) — compile-verified, matrix operand order checked by reading, not run on hardware
//
#include <hip/hip_runtime.h>
#include <math.h>

// ---------------------------------------------------------------------------
// Model constants (from reference)
// ---------------------------------------------------------------------------
#define Dm     512
#define Hh     64
#define HD     8
#define Ll     4
#define FFN    2048
#define RECS   3
#define Bb     2
#define Pp     256
#define Nn     512          // 2*P
#define C_IN   1088         // 2*D + H
#define TOK    (Bb*Nn)      // 1024 tokens
#define SCALE  0.35355339059327373f   // HD^-0.5

// Input indices: jax pytree flatten = sorted dict keys.
enum {
  IN_X = 0, IN_MASK = 1, IN_POS = 2,
  IN_B1CD = 3, IN_B2CD = 4, IN_BPC = 5,
  IN_B_EMB = 6, IN_B_FIN = 7, IN_G_EMB = 8, IN_G_FIN = 9,
  IN_L_B1 = 10, IN_L_B2 = 11, IN_L_BF1 = 12, IN_L_BF2 = 13,
  IN_L_BO = 14, IN_L_BQKV = 15, IN_L_G1 = 16, IN_L_G2 = 17,
  IN_L_W1 = 18, IN_L_W2 = 19, IN_L_WO = 20, IN_L_WQKV = 21,
  IN_W1CD = 22, IN_W2CD = 23, IN_WPC = 24
};

typedef __attribute__((ext_vector_type(16))) _Float16 v16h;
typedef __attribute__((ext_vector_type(8)))  float    v8f;
typedef unsigned int uint32x4 __attribute__((ext_vector_type(4)));
typedef int          int32x8  __attribute__((ext_vector_type(8)));
typedef int          int32x4  __attribute__((ext_vector_type(4)));

#if __has_builtin(__builtin_amdgcn_tensor_load_to_lds)
#define HAVE_TDM 1
#else
#define HAVE_TDM 0
#endif

// ---------------------------------------------------------------------------
// wave32 reductions
// ---------------------------------------------------------------------------
__device__ __forceinline__ float wave_sum(float v) {
  #pragma unroll
  for (int off = 16; off > 0; off >>= 1) v += __shfl_xor(v, off, 32);
  return v;
}
__device__ __forceinline__ float wave_max(float v) {
  #pragma unroll
  for (int off = 16; off > 0; off >>= 1) v = fmaxf(v, __shfl_xor(v, off, 32));
  return v;
}

// ---------------------------------------------------------------------------
// Stage a 2D tile [rows x d0] (row stride strideElems in global, ldsRowStride
// in LDS) into LDS. TDM path: one Tensor-Data-Mover descriptor issued by
// thread 0, LDS padding via D# pad fields (interval=256 dwords), completion
// tracked with TENSORcnt. Fallback: cooperative copy.
// Assumes blockDim.x == 256 and all threads reach this call.
// ---------------------------------------------------------------------------
template<typename T>
__device__ __forceinline__ void stage_tile(const T* __restrict__ g, T* lds,
                                           int d0, int rows, int strideElems,
                                           int ldsRowStride) {
  __syncthreads();                       // protect LDS from in-flight readers
#if HAVE_TDM
  if (threadIdx.x == 0) {
    unsigned long long ga = (unsigned long long)(uintptr_t)g;
    unsigned lds_off = (unsigned)(uintptr_t)lds;   // flat LDS aperture low bits
    uint32x4 g0;
    g0.x = 1u;                                          // count=1, user mode
    g0.y = lds_off;                                     // D#.lds_addr
    g0.z = (unsigned)ga;                                // global_addr[31:0]
    g0.w = (unsigned)((ga >> 32) & 0x1ffffffULL) | (2u << 30);  // [56:32]|type=2
    unsigned szcode = (sizeof(T) == 2) ? 1u : 2u;       // 2B / 4B elements
    unsigned w0 = (szcode << 16);                       // wg_mask=0, data_size
    if (ldsRowStride != d0) {                           // LDS anti-bank-conflict pad
      unsigned padDw = (unsigned)((ldsRowStride - d0) * (int)sizeof(T)) >> 2;
      w0 |= (1u << 20) | (7u << 22) | ((padDw - 1u) << 25);  // every 256 dwords
    }
    int32x8 g1;
    g1[0] = (int)w0;
    g1[1] = (int)(((unsigned)d0 & 0xffffu) << 16);           // tensor_dim0 lo
    g1[2] = (int)(((unsigned)d0 >> 16) | ((unsigned)rows << 16)); // dim0 hi|dim1 lo
    g1[3] = (int)((unsigned)d0 << 16);                       // dim1 hi=0 | tile_dim0
    g1[4] = (int)(unsigned)rows;                             // tile_dim1 | tile_dim2=0
    g1[5] = (int)(unsigned)strideElems;                      // dim0_stride lo
    g1[6] = 0; g1[7] = 0;                                    // stride hi / dim1_stride
    int32x4 z4 = {0, 0, 0, 0};
    int32x8 z8 = {0, 0, 0, 0, 0, 0, 0, 0};
    // 6-arg toolchain variant (clang-23 / therock headers)
    __builtin_amdgcn_tensor_load_to_lds(g0, g1, z4, z4, z8, 0);
  }
  if (threadIdx.x < 32) __builtin_amdgcn_s_wait_tensorcnt(0);
  __syncthreads();
#else
  for (int idx = threadIdx.x; idx < rows * d0; idx += 256) {
    int r = idx / d0, c0 = idx - r * d0;
    lds[r * ldsRowStride + c0] = g[(size_t)r * strideElems + c0];
  }
  __syncthreads();
#endif
}

// ---------------------------------------------------------------------------
// Weight convert: f32 [K x N] (leading dim ld, row offset ro) -> f16 [N x K]
// ---------------------------------------------------------------------------
__global__ __launch_bounds__(256) void convT_kernel(
    const float* __restrict__ W, _Float16* __restrict__ Wt,
    int K, int Nc, int ld, int ro) {
  int idx = blockIdx.x * 256 + threadIdx.x;
  if (idx >= K * Nc) return;
  int n = idx / K, k = idx - n * K;
  Wt[idx] = (_Float16)W[(size_t)(k + ro) * ld + n];
}

// ---------------------------------------------------------------------------
// WMMA GEMM: C[M x N] = A_f32[M x K] * Wt_f16[N x K]^T (+ biasvec) (+gelu)
// 256 thr = 8 waves -> 32x64 tile. B panel (64 x 512 halfs) staged via TDM
// into LDS with 522-half row stride (pad => conflict-free B-fragment loads).
// ROWMAP: 0 linear, 1 first-half tokens, 2 second-half tokens (dec1/dec2).
// ---------------------------------------------------------------------------
#define BP_STRIDE 522   // 512 + 10 halfs (20B pad per 1024B row)

template<int ROWMAP, int GELU>
__global__ __launch_bounds__(256) void gemm_wmma_kernel(
    const float* __restrict__ A, const _Float16* __restrict__ Wt,
    const float* __restrict__ biasvec, float* __restrict__ C,
    int M, int K, int Ncols) {
  __shared__ _Float16 smB[64 * BP_STRIDE];
  const int wave   = threadIdx.x >> 5;
  const int lane   = threadIdx.x & 31;
  const int l15    = lane & 15;
  const int half16 = lane >> 4;
  const int koff   = half16 * 8;
  const int m0 = blockIdx.y * 32 + (wave & 1) * 16;
  const int n0 = blockIdx.x * 64 + (wave >> 1) * 16;

  int mrow = m0 + l15;
  int arow = mrow;
  if (ROWMAP == 1) arow = (mrow >> 8) * Nn + (mrow & 255);
  if (ROWMAP == 2) arow = (mrow >> 8) * Nn + (mrow & 255) + Pp;
  const float* Arow = A + (size_t)arow * K;
  const _Float16* BL = smB + ((wave >> 1) * 16 + l15) * BP_STRIDE;
  const float bv = biasvec ? biasvec[n0 + l15] : 0.0f;   // hoisted epilogue bias

  v8f c = {};
  for (int k0 = 0; k0 < K; k0 += 512) {
    // async-DMA the B panel for this k-chunk into LDS
    stage_tile<_Float16>(Wt + (size_t)(blockIdx.x * 64) * K + k0,
                         smB, 512, 64, K, BP_STRIDE);
    for (int kk = 0; kk < 512; kk += 32) {
      if (kk + 32 < 512) __builtin_prefetch(Arow + k0 + kk + 32, 0, 3);
      v16h a, b;
      #pragma unroll
      for (int vg = 0; vg < 8; ++vg) {
        // e=2*vg -> K = e + (e>=8 ? 8:0) + (lane>=16 ? 8:0)  (ISA f16 layout)
        int kb = 2 * vg + ((vg >= 4) ? 8 : 0) + koff;
        float2 av = *(const float2*)(Arow + k0 + kk + kb);
        a[2*vg]     = (_Float16)av.x;
        a[2*vg + 1] = (_Float16)av.y;
        b[2*vg]     = BL[kk + kb];
        b[2*vg + 1] = BL[kk + kb + 1];
      }
      c = __builtin_amdgcn_wmma_f32_16x16x32_f16(false, a, false, b,
                                                 (short)0, c, false, false);
    }
  }
  #pragma unroll
  for (int r = 0; r < 8; ++r) {
    int m = m0 + r + half16 * 8;
    float v = c[r] + bv;
    if (GELU) v = 0.5f * v * (1.0f + erff(v * 0.70710678118654752f));
    C[(size_t)m * Ncols + (n0 + l15)] = v;
  }
}

// ---------------------------------------------------------------------------
// (Add +) LayerNorm over D=512, wave per token.
// ---------------------------------------------------------------------------
__global__ __launch_bounds__(256) void add_ln_kernel(
    const float* __restrict__ X, const float* __restrict__ R,
    const float* __restrict__ g, const float* __restrict__ bvec,
    const unsigned char* __restrict__ mask, float* __restrict__ Y) {
  int wave = threadIdx.x >> 5, lane = threadIdx.x & 31;
  int t = blockIdx.x * 8 + wave;                 // t < TOK
  const float* xr = X + (size_t)t * Dm;
  const float* rr = R ? R + (size_t)t * Dm : nullptr;
  float v[16]; float s = 0.f;
  #pragma unroll
  for (int u = 0; u < 16; ++u) {
    int d = u * 32 + lane;
    float x = xr[d]; if (rr) x += rr[d];
    v[u] = x; s += x;
  }
  float mean = wave_sum(s) * (1.0f / Dm);
  float var = 0.f;
  #pragma unroll
  for (int u = 0; u < 16; ++u) { float d = v[u] - mean; var += d * d; }
  var = wave_sum(var) * (1.0f / Dm);
  float rst = rsqrtf(var + 1e-5f);
  float mm = (mask && mask[t]) ? 0.0f : 1.0f;
  float* yr = Y + (size_t)t * Dm;
  #pragma unroll
  for (int u = 0; u < 16; ++u) {
    int d = u * 32 + lane;
    yr[d] = ((v[u] - mean) * rst * g[d] + bvec[d]) * mm;
  }
}

// ---------------------------------------------------------------------------
// Attention: one wave per (b,h,i). HD=8 -> VALU scores. Bias/score tensor is
// the bandwidth hog: read+rewritten in place with float4 (512B/instr/wave).
// ---------------------------------------------------------------------------
__global__ __launch_bounds__(256) void attn_kernel(
    const float* __restrict__ qkv, float* __restrict__ biasbuf,
    const unsigned char* __restrict__ mask, float* __restrict__ O) {
  int wave = threadIdx.x >> 5, lane = threadIdx.x & 31;
  int gid = blockIdx.x * 8 + wave;               // < B*H*N
  int b = gid >> 15;
  int h = (gid >> 9) & 63;
  int i = gid & 511;

  const float* qrow = qkv + ((size_t)(b * Nn + i)) * (3 * Dm) + h * HD;
  float q[HD];
  #pragma unroll
  for (int d = 0; d < HD; ++d) q[d] = qrow[d] * SCALE;

  float* srow = biasbuf + (((size_t)(b * Hh + h) * Nn + i)) * Nn;
  float s[16]; float mx = -1e30f;
  #pragma unroll
  for (int t = 0; t < 4; ++t) {
    int jb = t * 128 + lane * 4;
    float4 bv = *(const float4*)(srow + jb);
    uchar4 m4 = *(const uchar4*)(mask + b * Nn + jb);
    float sv[4];
    #pragma unroll
    for (int u = 0; u < 4; ++u) {
      const float* krow = qkv + ((size_t)(b * Nn + jb + u)) * (3 * Dm) + Dm + h * HD;
      float4 k0 = *(const float4*)(krow);
      float4 k1 = *(const float4*)(krow + 4);
      float dot = q[0]*k0.x + q[1]*k0.y + q[2]*k0.z + q[3]*k0.w
                + q[4]*k1.x + q[5]*k1.y + q[6]*k1.z + q[7]*k1.w;
      float bb = (u == 0) ? bv.x : (u == 1) ? bv.y : (u == 2) ? bv.z : bv.w;
      unsigned char mk = (u == 0) ? m4.x : (u == 1) ? m4.y : (u == 2) ? m4.z : m4.w;
      float x = dot + bb;
      if (mk) x = -1e30f;
      sv[u] = x; s[t * 4 + u] = x; mx = fmaxf(mx, x);
    }
    *(float4*)(srow + jb) = make_float4(sv[0], sv[1], sv[2], sv[3]);
  }
  mx = wave_max(mx);
  float sum = 0.f, acc[HD];
  #pragma unroll
  for (int d = 0; d < HD; ++d) acc[d] = 0.f;
  #pragma unroll
  for (int t = 0; t < 4; ++t) {
    int jb = t * 128 + lane * 4;
    #pragma unroll
    for (int u = 0; u < 4; ++u) {
      float p = __expf(s[t * 4 + u] - mx); sum += p;
      const float* vrow = qkv + ((size_t)(b * Nn + jb + u)) * (3 * Dm) + 2 * Dm + h * HD;
      float4 v0 = *(const float4*)(vrow);
      float4 v1 = *(const float4*)(vrow + 4);
      acc[0] += p * v0.x; acc[1] += p * v0.y; acc[2] += p * v0.z; acc[3] += p * v0.w;
      acc[4] += p * v1.x; acc[5] += p * v1.y; acc[6] += p * v1.z; acc[7] += p * v1.w;
    }
  }
  float inv = 1.0f / wave_sum(sum);
  #pragma unroll
  for (int d = 0; d < HD; ++d) {
    float tot = wave_sum(acc[d]);
    if (lane == d) O[(size_t)(b * Nn + i) * Dm + h * HD + d] = tot * inv;
  }
}

// ---------------------------------------------------------------------------
// pair-rep dot products: a[t]=x[t]·w_pc[:D], b[t]=x[t]·w_pc[D:], wave/token
// ---------------------------------------------------------------------------
__global__ __launch_bounds__(256) void dotpc_kernel(
    const float* __restrict__ X, const float* __restrict__ wpc,
    float* __restrict__ aD, float* __restrict__ bD) {
  int wave = threadIdx.x >> 5, lane = threadIdx.x & 31;
  int t = blockIdx.x * 8 + wave;
  const float* row = X + (size_t)t * Dm;
  float sa = 0.f, sb = 0.f;
  #pragma unroll
  for (int u = 0; u < 16; ++u) {
    int d = u * 32 + lane;
    float x = row[d];
    sa += x * wpc[d]; sb += x * wpc[Dm + d];
  }
  sa = wave_sum(sa); sb = wave_sum(sb);
  if (lane == 0) { aD[t] = sa; bD[t] = sb; }
}

// bias[b,h,i,j] = same-half ? a[i]+b[j]+b_pc : 0, float4 per thread, x64 heads
__global__ __launch_bounds__(256) void bias_init_kernel(
    const float* __restrict__ aD, const float* __restrict__ bD,
    const float* __restrict__ bpc, float* __restrict__ biasbuf) {
  int idx = blockIdx.x * 256 + threadIdx.x;      // < B*N*N/4
  int b = idx >> 16;                              // N*N/4 = 65536
  int rem = idx & 65535;
  int i = rem >> 7, j4 = (rem & 127) * 4;
  bool same = (i >> 8) == (j4 >> 8);
  float ai = aD[b * Nn + i] + bpc[0];
  float4 v;
  v.x = same ? ai + bD[b * Nn + j4 + 0] : 0.0f;
  v.y = same ? ai + bD[b * Nn + j4 + 1] : 0.0f;
  v.z = same ? ai + bD[b * Nn + j4 + 2] : 0.0f;
  v.w = same ? ai + bD[b * Nn + j4 + 3] : 0.0f;
  size_t base = (size_t)b * Hh * Nn * Nn + (size_t)i * Nn + j4;
  for (int h = 0; h < Hh; ++h)
    *(float4*)(biasbuf + base + (size_t)h * Nn * Nn) = v;
}

__global__ void init_kernel(float* accum) {
  if (threadIdx.x == 0) accum[0] = 0.0f;
}

// ---------------------------------------------------------------------------
// Decoder: block = (b, i, 32-wide j block); 256 thr = 32 j x 8 c-lanes.
// w1cd[:H] (64x1088 f32) is TDM-staged into LDS in 4 chunks of 272 columns:
// kills the ~1TB of repeated L2 reads; inner loop is pure LDS FMA.
// ---------------------------------------------------------------------------
__global__ __launch_bounds__(256) void decode_kernel(
    const float* __restrict__ biasbuf, const float* __restrict__ r1,
    const float* __restrict__ r2, const float* __restrict__ w1cd,
    const float* __restrict__ w2cd, const float* __restrict__ b2cd,
    const float* __restrict__ pos, const unsigned char* __restrict__ mask,
    float* __restrict__ accum) {
  __shared__ float w1s[64 * 272];  // 69.6 KB chunk of w1cd
  __shared__ float ppd[32][68];    // padded: avoid 4-way bank conflicts
  __shared__ float part[32][8];
  __shared__ float bsum;
  int jblk = blockIdx.x, i = blockIdx.y, b = blockIdx.z;
  int tid = threadIdx.x;
  if (tid == 0) bsum = 0.0f;

  for (int e = tid; e < 32 * Hh; e += 256) {
    int jl = e >> 6, h = e & 63;
    int j = jblk * 32 + jl;
    size_t base = (size_t)(b * Hh + h) * Nn * Nn;
    float p1 = biasbuf[base + (size_t)i * Nn + (Pp + j)];
    float p2 = biasbuf[base + (size_t)(Pp + j) * Nn + i];
    float v = 0.5f * (p1 + p2);
    if (v < -1e29f) v = 0.0f;                    // isneginf guard
    ppd[jl][h] = v;
  }

  int jl = tid >> 3, cl = tid & 7;
  int j = jblk * 32 + jl;
  const float* r1row = r1 + (size_t)(b * Pp + i) * C_IN;
  const float* r2row = r2 + (size_t)(b * Pp + j) * C_IN;
  float acc = 0.f;
  for (int ch = 0; ch < 4; ++ch) {
    stage_tile<float>(w1cd + ch * 272, w1s, 272, 64, C_IN, 272);
    for (int k = 0; k < 34; ++k) {
      int cc = cl + 8 * k;
      int c = ch * 272 + cc;
      float u = r1row[c] + r2row[c];
      #pragma unroll 8
      for (int h = 0; h < Hh; ++h) u += ppd[jl][h] * w1s[h * 272 + cc];
      acc += fmaxf(u, 0.0f) * w2cd[c];
    }
  }
  part[jl][cl] = acc;
  __syncthreads();

  if (cl == 0) {
    float pred = b2cd[0];
    #pragma unroll
    for (int t = 0; t < 8; ++t) pred += part[jl][t];
    pred = (pred > 0.f) ? (pred + 1.0f) : __expf(pred);   // elu(x)+1
    const float* ca1 = pos + (((size_t)(b * Nn + i)) * 4 + 1) * 3;
    const float* ca2 = pos + (((size_t)(b * Nn + Pp + j)) * 4 + 1) * 3;
    float dx = ca1[0] - ca2[0], dy = ca1[1] - ca2[1], dz = ca1[2] - ca2[2];
    float dist = sqrtf(dx * dx + dy * dy + dz * dz);
    float e = pred - dist;
    bool valid = !(mask[b * Nn + i] || mask[b * Nn + Pp + j]);
    atomicAdd(&bsum, valid ? e * e : 0.0f);
  }
  __syncthreads();
  if (tid == 0) atomicAdd(&accum[0], bsum);
}

__global__ __launch_bounds__(256) void finalize_kernel(
    const float* __restrict__ accum, const unsigned char* __restrict__ mask,
    float* __restrict__ out) {
  __shared__ float cnt[256];
  int tid = threadIdx.x;
  float c = 0.f;
  for (int idx = tid; idx < Bb * Pp * Pp; idx += 256) {
    int b = idx >> 16, i = (idx >> 8) & 255, j = idx & 255;
    if (!(mask[b * Nn + i] || mask[b * Nn + Pp + j])) c += 1.0f;
  }
  cnt[tid] = c; __syncthreads();
  for (int s = 128; s > 0; s >>= 1) {
    if (tid < s) cnt[tid] += cnt[tid + s];
    __syncthreads();
  }
  if (tid == 0) out[0] = accum[0] / fmaxf(cnt[0], 1.0f);
}

// ---------------------------------------------------------------------------
// Host orchestration
// ---------------------------------------------------------------------------
extern "C" void kernel_launch(void* const* d_in, const int* in_sizes, int n_in,
                              void* d_out, int out_size, void* d_ws, size_t ws_size,
                              hipStream_t stream) {
  const float* x_in  = (const float*)d_in[IN_X];
  const unsigned char* mask = (const unsigned char*)d_in[IN_MASK];
  const float* pos   = (const float*)d_in[IN_POS];
  const float* b1cd  = (const float*)d_in[IN_B1CD];
  const float* b2cd  = (const float*)d_in[IN_B2CD];
  const float* bpc   = (const float*)d_in[IN_BPC];
  const float* b_emb = (const float*)d_in[IN_B_EMB];
  const float* b_fin = (const float*)d_in[IN_B_FIN];
  const float* g_emb = (const float*)d_in[IN_G_EMB];
  const float* g_fin = (const float*)d_in[IN_G_FIN];
  const float* Lb1   = (const float*)d_in[IN_L_B1];
  const float* Lb2   = (const float*)d_in[IN_L_B2];
  const float* Lbf1  = (const float*)d_in[IN_L_BF1];
  const float* Lbf2  = (const float*)d_in[IN_L_BF2];
  const float* Lbo   = (const float*)d_in[IN_L_BO];
  const float* Lbqkv = (const float*)d_in[IN_L_BQKV];
  const float* Lg1   = (const float*)d_in[IN_L_G1];
  const float* Lg2   = (const float*)d_in[IN_L_G2];
  const float* Lw1   = (const float*)d_in[IN_L_W1];
  const float* Lw2   = (const float*)d_in[IN_L_W2];
  const float* Lwo   = (const float*)d_in[IN_L_WO];
  const float* Lwqkv = (const float*)d_in[IN_L_WQKV];
  const float* w1cd  = (const float*)d_in[IN_W1CD];
  const float* w2cd  = (const float*)d_in[IN_W2CD];
  const float* wpc   = (const float*)d_in[IN_WPC];

  // ---- workspace carve --------------------------------------------------
  char* base = (char*)d_ws; size_t off = 0;
  auto carve = [&](size_t bytes) -> char* {
    char* p = base + off; off += (bytes + 255) & ~(size_t)255; return p;
  };
  float* biasbuf = (float*)carve((size_t)Bb * Hh * Nn * Nn * 4); // 134 MB
  float* x    = (float*)carve((size_t)TOK * Dm * 4);
  float* qkv  = (float*)carve((size_t)TOK * 3 * Dm * 4);
  float* o    = (float*)carve((size_t)TOK * Dm * 4);
  float* tmp  = (float*)carve((size_t)TOK * Dm * 4);
  float* hbuf = (float*)carve((size_t)TOK * FFN * 4);
  float* aD   = (float*)carve(TOK * 4);
  float* bD   = (float*)carve(TOK * 4);
  float* r1   = (float*)carve((size_t)Bb * Pp * C_IN * 4);
  float* r2   = (float*)carve((size_t)Bb * Pp * C_IN * 4);
  float* accum = (float*)carve(256);
  _Float16* wqkvT = (_Float16*)carve((size_t)Ll * Dm * 3 * Dm * 2);
  _Float16* woT   = (_Float16*)carve((size_t)Ll * Dm * Dm * 2);
  _Float16* w1T   = (_Float16*)carve((size_t)Ll * Dm * FFN * 2);
  _Float16* w2T   = (_Float16*)carve((size_t)Ll * FFN * Dm * 2);
  _Float16* w1cdT1 = (_Float16*)carve((size_t)Dm * C_IN * 2);
  _Float16* w1cdT2 = (_Float16*)carve((size_t)Dm * C_IN * 2);

  auto conv = [&](const float* W, _Float16* Wt, int K, int Nc, int ld, int ro) {
    int elems = K * Nc;
    convT_kernel<<<(elems + 255) / 256, 256, 0, stream>>>(W, Wt, K, Nc, ld, ro);
  };
  // ---- once-per-launch weight conversion (deterministic) ----------------
  for (int l = 0; l < Ll; ++l) {
    conv(Lwqkv + (size_t)l * Dm * 3 * Dm, wqkvT + (size_t)l * Dm * 3 * Dm, Dm, 3 * Dm, 3 * Dm, 0);
    conv(Lwo   + (size_t)l * Dm * Dm,     woT   + (size_t)l * Dm * Dm,     Dm, Dm,     Dm,     0);
    conv(Lw1   + (size_t)l * Dm * FFN,    w1T   + (size_t)l * Dm * FFN,    Dm, FFN,    FFN,    0);
    conv(Lw2   + (size_t)l * FFN * Dm,    w2T   + (size_t)l * FFN * Dm,    FFN, Dm,    Dm,     0);
  }
  conv(w1cd, w1cdT1, Dm, C_IN, C_IN, Hh);        // rows H..H+D
  conv(w1cd, w1cdT2, Dm, C_IN, C_IN, Hh + Dm);   // rows H+D..H+2D

  // ---- pair-rep bias init ----------------------------------------------
  init_kernel<<<1, 32, 0, stream>>>(accum);
  dotpc_kernel<<<TOK / 8, 256, 0, stream>>>(x_in, wpc, aD, bD);
  bias_init_kernel<<<(Bb * Nn * Nn / 4) / 256, 256, 0, stream>>>(aD, bD, bpc, biasbuf);

  // ---- REC x L encoder --------------------------------------------------
  const float* src = x_in;
  for (int rec = 0; rec < RECS; ++rec) {
    add_ln_kernel<<<TOK / 8, 256, 0, stream>>>(src, nullptr, g_emb, b_emb, mask, x);
    src = x;
    for (int l = 0; l < Ll; ++l) {
      gemm_wmma_kernel<0, 0><<<dim3(3 * Dm / 64, TOK / 32), 256, 0, stream>>>(
          x, wqkvT + (size_t)l * Dm * 3 * Dm, Lbqkv + l * 3 * Dm, qkv, TOK, Dm, 3 * Dm);
      attn_kernel<<<(Bb * Hh * Nn) / 8, 256, 0, stream>>>(qkv, biasbuf, mask, o);
      gemm_wmma_kernel<0, 0><<<dim3(Dm / 64, TOK / 32), 256, 0, stream>>>(
          o, woT + (size_t)l * Dm * Dm, Lbo + l * Dm, tmp, TOK, Dm, Dm);
      add_ln_kernel<<<TOK / 8, 256, 0, stream>>>(x, tmp, Lg1 + l * Dm, Lb1 + l * Dm, nullptr, x);
      gemm_wmma_kernel<0, 1><<<dim3(FFN / 64, TOK / 32), 256, 0, stream>>>(
          x, w1T + (size_t)l * Dm * FFN, Lbf1 + l * FFN, hbuf, TOK, Dm, FFN);
      gemm_wmma_kernel<0, 0><<<dim3(Dm / 64, TOK / 32), 256, 0, stream>>>(
          hbuf, w2T + (size_t)l * FFN * Dm, Lbf2 + l * Dm, tmp, TOK, FFN, Dm);
      add_ln_kernel<<<TOK / 8, 256, 0, stream>>>(x, tmp, Lg2 + l * Dm, Lb2 + l * Dm, nullptr, x);
    }
    add_ln_kernel<<<TOK / 8, 256, 0, stream>>>(x, nullptr, g_fin, b_fin, nullptr, x);
  }

  // ---- decoder ----------------------------------------------------------
  gemm_wmma_kernel<1, 0><<<dim3(C_IN / 64, (Bb * Pp) / 32), 256, 0, stream>>>(
      x, w1cdT1, b1cd, r1, Bb * Pp, Dm, C_IN);          // dec1, b1_cd folded in
  gemm_wmma_kernel<2, 0><<<dim3(C_IN / 64, (Bb * Pp) / 32), 256, 0, stream>>>(
      x, w1cdT2, nullptr, r2, Bb * Pp, Dm, C_IN);       // dec2
  decode_kernel<<<dim3(Pp / 32, Pp, Bb), 256, 0, stream>>>(
      biasbuf, r1, r2, w1cd, w2cd, b2cd, pos, mask, accum);
  finalize_kernel<<<1, 256, 0, stream>>>(accum, mask, (float*)d_out);
}
